// Modulated2d_71064528880342
// MI455X (gfx1250) — matrix operand order repaired and checked by reading, not compiled, not run on hardware
//
#include <hip/hip_runtime.h>
#include <hip/hip_bf16.h>

// ---------------------------------------------------------------------------
// Modulated grouped conv (StyleGAN2) as per-batch implicit GEMM on CDNA5 WMMA.
// B=16, CIN=512, COUT=512, H=W=64, K=3.  GEMM per batch: M=512, N=4096, K=4608.
// bf16 operands, f32 accumulation via v_wmma_f32_16x16x32_bf16.
//
// v3: A-tile staged with global_load_async_to_lds_b128 (ASYNCcnt path,
//     s_wait_asynccnt before the publishing barrier); zero-halo padded bf16
//     input; natural-order LDS A layout (contiguous 32B fragments);
//     128x256 block tile / 64x64 wave tile; K-stage depth 64
//     => 32 WMMAs per wave per barrier pair.
// ---------------------------------------------------------------------------

#define Bn   16
#define CIN  512
#define COUT 512
#define Hn   64
#define Wn   64
#define HW   (Hn * Wn)          // 4096
#define KK   9
#define KTOT (CIN * KK)         // 4608
#define SCALE_F (1.0f / 67.88225099390856f)  // 1/sqrt(512*9)
#define EPS_F 1e-8f

// padded input geometry: [b][i][66][72], data at rows 1..64, cols 1..64
#define HP 66
#define WP 72
#define HWP (HP * WP)           // 4752

#define LPAD 72                 // LDS row length in bf16 for K-stage of 64 (+8 pad)
#define LDSA_ELEMS (128 * LPAD) // 9216 bf16 = 18432 B
#define LDSB_ELEMS (256 * LPAD) // 18432 bf16 = 36864 B

typedef __attribute__((ext_vector_type(16))) __bf16 bf16x16;
typedef __attribute__((ext_vector_type(8)))  __bf16 bf16x8;
typedef __attribute__((ext_vector_type(8)))  float  f32x8;

// ---------------------------------------------------------------------------
// CDNA5 async global->LDS copy (16B per lane), tracked by ASYNCcnt.
// lds_off is a byte offset from the wave's LDS allocation base.
// ---------------------------------------------------------------------------
static __device__ __forceinline__ void async_load_b128(unsigned lds_off, const void* gaddr) {
    asm volatile("global_load_async_to_lds_b128 %0, %1, off"
                 :: "v"(lds_off), "v"(gaddr)
                 : "memory");
}
static __device__ __forceinline__ void wait_async0() {
    asm volatile("s_wait_asynccnt 0" ::: "memory");
}

// ---------------------------------------------------------------------------
// Kernel A: x (f32, [b][i][64][64]) -> xpad (bf16, [b][i][66][72], zero halo)
// ---------------------------------------------------------------------------
__global__ __launch_bounds__(256) void xpad_kernel(const float* __restrict__ x,
                                                   __bf16* __restrict__ xp, int n) {
    int idx = blockIdx.x * 256 + threadIdx.x;
    if (idx >= n) return;
    int wp = idx % WP;
    int t  = idx / WP;
    int hp = t % HP;
    int bi = t / HP;                       // fused (b*512 + i)
    float v = 0.0f;
    int h = hp - 1, w = wp - 1;
    if (h >= 0 && h < Hn && w >= 0 && w < Wn)
        v = x[(size_t)bi * HW + h * Wn + w];
    xp[idx] = (__bf16)v;
}

// ---------------------------------------------------------------------------
// Kernel B: per (b,o) modulate + demodulate, emit bf16 weights in GEMM K-order:
//   wtsb[b][o][ (kh*3+kw)*512 + i ]
// ---------------------------------------------------------------------------
__global__ __launch_bounds__(256) void modw_kernel(const float* __restrict__ w,
                                                   const float* __restrict__ y,
                                                   __bf16* __restrict__ wtsb) {
    const int o = blockIdx.x;          // 0..511
    const int b = blockIdx.y;          // 0..15
    const int t = threadIdx.x;         // 0..255

    const float* wrow = w + (size_t)o * KTOT;   // [i][kh][kw] contiguous
    const float* yrow = y + (size_t)b * CIN;

    float vals[18];
    float ss = 0.0f;
#pragma unroll
    for (int u = 0; u < 18; ++u) {
        int j = t + 256 * u;           // 0..4607, j = i*9 + q  (coalesced read)
        int i = j / 9;
        float s = SCALE_F * wrow[j] * (yrow[i] + 1.0f);
        vals[u] = s;
        ss += s * s;
    }

    __shared__ float red[256];
    red[t] = ss;
    __syncthreads();
#pragma unroll
    for (int stv = 128; stv > 0; stv >>= 1) {
        if (t < stv) red[t] += red[t + stv];
        __syncthreads();
    }
    const float d = rsqrtf(red[0] + EPS_F);

    __bf16* outrow = wtsb + ((size_t)b * COUT + o) * KTOT;
#pragma unroll
    for (int u = 0; u < 18; ++u) {
        int j = t + 256 * u;
        int i = j / 9, q = j - i * 9;
        outrow[q * CIN + i] = (__bf16)(vals[u] * d);
    }
}

// ---------------------------------------------------------------------------
// Contiguous 32B LDS fragment read (two ds_load_b128, no repack).
// ---------------------------------------------------------------------------
static __device__ __forceinline__ bf16x16 ldsv32(const __bf16* p) {
    bf16x8 lo = *(const bf16x8*)p;
    bf16x8 hi = *(const bf16x8*)(p + 8);
    return __builtin_shufflevector(lo, hi,
                                   0, 1, 2, 3, 4, 5, 6, 7,
                                   8, 9, 10, 11, 12, 13, 14, 15);
}

// ---------------------------------------------------------------------------
// Kernel C: per-batch implicit GEMM with WMMA.
//   grid = (N/256 = 16, M/128 = 4, B = 16), block = 256 threads (8 wave32).
//   Block tile : 128 O x 256 pixels (4 full image rows).
//   Wave tile  : 64 O x 64 pixels = 4x4 accumulators of 16x16.
//   K stage    : 64 deep (two 32-K WMMA chunks) => 32 WMMAs / wave / barrier.
//
// LDS A row layout per 32-K chunk: K-groups of 8 stored in natural order {0,1,2,3}
// so lane ksel's A fragment (K = 16*ksel..16*ksel+15)
// is 16 contiguous elements at offset chunk*32 + ksel*16.
// LDS B layout is natural: lane ksel's fragment = 16 contiguous K at ksel*16.
// A tile is staged by the Tensor/async path (global_load_async_to_lds_b128).
// ---------------------------------------------------------------------------
__global__ __launch_bounds__(256) void conv_wmma_kernel(const __bf16* __restrict__ xp,
                                                        const __bf16* __restrict__ wtsb,
                                                        const float* __restrict__ bias,
                                                        float* __restrict__ out) {
    const int b   = blockIdx.z;
    const int o0  = blockIdx.y * 128;
    const int px0 = blockIdx.x * 256;
    const int h0  = px0 >> 6;                 // tile spans image rows h0..h0+3

    const int tid  = threadIdx.x;
    const int lane = tid & 31;
    const int wv   = tid >> 5;                // 0..7
    const int wM   = wv >> 2;                 // 0..1  (64 O rows each)
    const int wN   = wv & 3;                  // 0..3  (64 pixels each)
    const int m16  = lane & 15;
    const int ksel = lane >> 4;               // 0/1: lane half selects K-half

    // single shared block: ldsA at byte offset 0, ldsB after it
    __shared__ __align__(16) __bf16 ldsAll[LDSA_ELEMS + LDSB_ELEMS];  // 55296 B
    __bf16* const ldsA = ldsAll;
    __bf16* const ldsB = ldsAll + LDSA_ELEMS;

    f32x8 acc[4][4];
#pragma unroll
    for (int mi = 0; mi < 4; ++mi)
#pragma unroll
        for (int ni = 0; ni < 4; ++ni) acc[mi][ni] = (f32x8)0.0f;

    const __bf16* wb  = wtsb + ((size_t)b * COUT + o0) * KTOT;
    const __bf16* xpb = xp + (size_t)b * CIN * HWP;

    // A staging task for this thread: 4 x 16B chunks, natural-order dest
    // task = tid + 256*u : row = task>>3 (0..127), g = task&7 (K-group of 8)
    unsigned aLdsOff[4];
    size_t   aSrcOff[4];
#pragma unroll
    for (int u = 0; u < 4; ++u) {
        int task = tid + 256 * u;
        int row  = task >> 3;
        int g    = task & 7;
        int c    = g >> 2;
        int gg   = g & 3;
        int ggp  = gg;                                       // natural group order
        aLdsOff[u] = (unsigned)(row * LPAD + c * 32 + ggp * 8) * (unsigned)sizeof(__bf16);
        aSrcOff[u] = (size_t)row * KTOT + g * 8;             // + kbase per stage
    }

    // this thread owns pixel pxl = tid of the 256-pixel tile for B staging
    const int ph = tid >> 6, pw = tid & 63;
    const int rowcol = (h0 + ph) * WP + pw;   // padded row/col base (halo = +1 built in)

    for (int p = 0; p < KK; ++p) {
        const int kh = p / 3;                 // 0..2 (maps to -1..1 via halo)
        const int kw = p - 3 * kh;            // 0..2
        const __bf16* srcB0 = xpb + rowcol + kh * WP + kw;

        for (int s = 0; s < CIN / 64; ++s) {  // 8 K-stages of 64 per tap
            const int kbase = p * CIN + s * 64;
            const int i0    = s * 64;

            __syncthreads();   // protect LDS reuse from previous stage

            // ---- stage A: async global->LDS, 16B per lane, ASYNCcnt-tracked ----
#pragma unroll
            for (int u = 0; u < 4; ++u)
                async_load_b128(aLdsOff[u], wb + aSrcOff[u] + kbase);

            // ---- stage B: this thread's pixel, 64 i-values (regular stride) ----
            {
                const __bf16* srcB = srcB0 + (size_t)i0 * HWP;
                __bf16* dstB = &ldsB[tid * LPAD];
#pragma unroll
                for (int il = 0; il < 64; ++il)
                    dstB[il] = srcB[(size_t)il * HWP];
            }

            wait_async0();     // A-tile async copies complete (this wave)
            __syncthreads();   // publish A+B tiles to all waves

            // ---- two 32-K WMMA chunks ----
#pragma unroll
            for (int c = 0; c < 2; ++c) {
                bf16x16 af[4], bfg[4];
#pragma unroll
                for (int mi = 0; mi < 4; ++mi) {
                    int row = wM * 64 + mi * 16 + m16;
                    af[mi] = ldsv32(&ldsA[row * LPAD + c * 32 + ksel * 16]);
                }
#pragma unroll
                for (int ni = 0; ni < 4; ++ni) {
                    int col = wN * 64 + ni * 16 + m16;
                    bfg[ni] = ldsv32(&ldsB[col * LPAD + c * 32 + ksel * 16]);
                }
#pragma unroll
                for (int mi = 0; mi < 4; ++mi)
#pragma unroll
                    for (int ni = 0; ni < 4; ++ni)
                        acc[mi][ni] = __builtin_amdgcn_wmma_f32_16x16x32_bf16(
                            false, af[mi], false, bfg[ni],
                            (short)0, acc[mi][ni], false, false);
            }
        }
    }

    // ---- epilogue: C/D layout M = 8*ksel + r, N = m16; add bias; f32 store ----
#pragma unroll
    for (int mi = 0; mi < 4; ++mi) {
#pragma unroll
        for (int ni = 0; ni < 4; ++ni) {
            int colg = px0 + wN * 64 + ni * 16 + m16;
#pragma unroll
            for (int r = 0; r < 8; ++r) {
                int og = o0 + wM * 64 + mi * 16 + ksel * 8 + r;
                out[((size_t)b * COUT + og) * HW + colg] = acc[mi][ni][r] + bias[og];
            }
        }
    }
}

// ---------------------------------------------------------------------------
// Launch
// ---------------------------------------------------------------------------
extern "C" void kernel_launch(void* const* d_in, const int* in_sizes, int n_in,
                              void* d_out, int out_size, void* d_ws, size_t ws_size,
                              hipStream_t stream) {
    const float* x    = (const float*)d_in[0];   // [16,512,64,64]
    const float* y    = (const float*)d_in[1];   // [16,512]
    const float* w    = (const float*)d_in[2];   // [512,512,3,3]
    const float* bias = (const float*)d_in[3];   // [512]
    float* out = (float*)d_out;                  // [16,512,64,64]

    // Workspace: xpad bf16 [16][512][66][72] | wtsb bf16 [16][512][4608]
    const size_t xpad_elems = (size_t)Bn * CIN * HWP;      // 38,928,384
    __bf16* xpad = (__bf16*)d_ws;
    __bf16* wtsb = (__bf16*)((char*)d_ws + xpad_elems * sizeof(__bf16));

    const int np = (int)xpad_elems;
    xpad_kernel<<<(np + 255) / 256, 256, 0, stream>>>(x, xpad, np);

    modw_kernel<<<dim3(COUT, Bn), 256, 0, stream>>>(w, y, wtsb);

    conv_wmma_kernel<<<dim3(HW / 256, COUT / 128, Bn), 256, 0, stream>>>(xpad, wtsb, bias, out);
}
